// SpaceTimeTransformer_39754217291954
// MI455X (gfx1250) — matrix operand order repaired and checked by reading, not compiled, not run on hardware
//
#include <hip/hip_runtime.h>
#include <cstdint>
#include <cstddef>

// ---------------------------------------------------------------------------
// Types for CDNA5 WMMA (gfx1250, wave32)
// ---------------------------------------------------------------------------
typedef __bf16 bf16;
typedef __attribute__((ext_vector_type(16))) __bf16 v16bf;
typedef __attribute__((ext_vector_type(8)))  __bf16 v8bf;
typedef __attribute__((ext_vector_type(8)))  float  v8f;

#define DEV_ static __device__ __forceinline__

// ---------------------------------------------------------------------------
// Problem constants (SpaceTimeTransformer, B=4, F=8, 224x224, patch16, D=768)
// ---------------------------------------------------------------------------
constexpr int kB    = 4;
constexpr int kF    = 8;
constexpr int kNP   = 196;           // patches per frame (14x14)
constexpr int kN1   = 1569;          // 1 + F*n
constexpr int kH    = 12;            // heads
constexpr int kD    = 768;
constexpr int kDF   = 3072;
constexpr int kT    = kB * kN1;      // 6276 tokens
constexpr int kTp   = 6288;          // padded to multiple of 16
constexpr int kPT   = kB * kF * kNP; // 6272 patch tokens (multiple of 64)

constexpr size_t SZ_PATCHW = (size_t)768 * 768;     // 589824
constexpr size_t SZ_QKVW   = (size_t)2304 * 768;    // 1769472
constexpr size_t SZ_PROJW  = (size_t)768 * 768;
constexpr size_t SZ_W1     = (size_t)3072 * 768;
constexpr size_t SZ_W2     = (size_t)768 * 3072;
constexpr size_t SZ_LAYER  = 2 * SZ_QKVW + 2 * SZ_PROJW + SZ_W1 + SZ_W2;   // 9437184
constexpr size_t SZ_HEADP  = (size_t)1008 * 768;    // head weight padded to 1008 rows

// ---------------------------------------------------------------------------
// WMMA fragment loaders (bf16, 16x16x32) — contiguous k order (k = 16h + s) in A and B alike.
// A (16x32, MxK): lanes 0-15 = rows 0-15 with K[0..15] contiguous; lanes 16-31
//   same rows with K[16..31].
// B (32x16, KxN): lanes 0-15 = col n, K[0..15] contiguous; lanes 16-31 same
//   col with K[16..31]. Both operands are K-contiguous for row-major (.,K).
// ---------------------------------------------------------------------------
DEV_ v16bf load_fragA(const bf16* __restrict__ base, int ld) {
  const int lane = threadIdx.x & 31;
  const int r = lane & 15;
  const int h = lane >> 4;
  const bf16* p = base + (size_t)r * ld + h * 16;
  v8bf lo = *(const v8bf*)(p);
  v8bf hi = *(const v8bf*)(p + 8);
  return __builtin_shufflevector(lo, hi, 0, 1, 2, 3, 4, 5, 6, 7,
                                         8, 9, 10, 11, 12, 13, 14, 15);
}

DEV_ v16bf load_fragB(const bf16* __restrict__ base, int ld) {
  const int lane = threadIdx.x & 31;
  const int r = lane & 15;
  const int h = lane >> 4;
  const bf16* p = base + (size_t)r * ld + h * 16;
  return *(const v16bf*)p;   // 32B, compiler splits into 2x b128
}

// ---------------------------------------------------------------------------
// Generic WMMA GEMM:  out[m,n] = sum_k A[m,k] * W[n,k]  (+bias) (+resid) (+gelu)
//   A: Mp x K bf16 row-major (Mp multiple of 16)
//   W: Npad x K bf16 row-major (Npad multiple of 16, zero-padded rows ok)
//   stores guarded to col < Nlog, row < Mp; out leading dim = Nlog
// Per-wave tile 32x64 (2x4 frags); block = 8 waves -> 64x256 tile.
// K-loop unrolled x2 (K is always a multiple of 64) -> 16 WMMA / iteration.
// MODE: 0 = fp32 out, 1 = fp32 out + residual, 2 = bf16 out + exact GELU.
// ---------------------------------------------------------------------------
template <int MODE>
__global__ __launch_bounds__(256) void wmma_gemm_t(
    const bf16* __restrict__ A, const bf16* __restrict__ W,
    const float* __restrict__ bias, const float* __restrict__ resid,
    float* __restrict__ outf, bf16* __restrict__ outb,
    int Mp, int Npad, int Nlog, int K)
{
  const int wave = threadIdx.x >> 5;
  const int wm = wave >> 2;           // 0..1
  const int wn = wave & 3;            // 0..3
  const int mb0 = blockIdx.y * 64 + wm * 32;
  const int nb0 = blockIdx.x * 256 + wn * 64;

  v8f acc[2][4];
  const v8f vzero = {0.f, 0.f, 0.f, 0.f, 0.f, 0.f, 0.f, 0.f};
#pragma unroll
  for (int i = 0; i < 2; ++i)
#pragma unroll
    for (int j = 0; j < 4; ++j) acc[i][j] = vzero;

  int ma[2], nb[4];
#pragma unroll
  for (int i = 0; i < 2; ++i) { int m = mb0 + 16 * i; ma[i] = (m + 16 <= Mp)   ? m : 0; }
#pragma unroll
  for (int j = 0; j < 4; ++j) { int n = nb0 + 16 * j; nb[j] = (n + 16 <= Npad) ? n : 0; }

  const bf16* pa[2];
  const bf16* pb[4];
#pragma unroll
  for (int i = 0; i < 2; ++i) pa[i] = A + (size_t)ma[i] * K;
#pragma unroll
  for (int j = 0; j < 4; ++j) pb[j] = W + (size_t)nb[j] * K;

  for (int k0 = 0; k0 < K; k0 += 64) {
    // -- first 32-deep step
    v16bf a0[2], b0[4];
#pragma unroll
    for (int i = 0; i < 2; ++i) a0[i] = load_fragA(pa[i] + k0, K);
#pragma unroll
    for (int j = 0; j < 4; ++j) b0[j] = load_fragB(pb[j] + k0, K);
    // -- second 32-deep step (independent load group, overlaps first WMMAs)
    v16bf a1[2], b1[4];
#pragma unroll
    for (int i = 0; i < 2; ++i) a1[i] = load_fragA(pa[i] + k0 + 32, K);
#pragma unroll
    for (int j = 0; j < 4; ++j) b1[j] = load_fragB(pb[j] + k0 + 32, K);

#pragma unroll
    for (int i = 0; i < 2; ++i)
#pragma unroll
      for (int j = 0; j < 4; ++j)
        acc[i][j] = __builtin_amdgcn_wmma_f32_16x16x32_bf16(
            false, a0[i], false, b0[j], (short)0, acc[i][j], false, false);
#pragma unroll
    for (int i = 0; i < 2; ++i)
#pragma unroll
      for (int j = 0; j < 4; ++j)
        acc[i][j] = __builtin_amdgcn_wmma_f32_16x16x32_bf16(
            false, a1[i], false, b1[j], (short)0, acc[i][j], false, false);
  }

  // 16x16 f32 C/D layout: lane n holds col n (lanes 16-31: col n-16, rows +8),
  // vgpr e holds row e (+8 for upper half).
  const int lane = threadIdx.x & 31;
  const int colL = lane & 15;
  const int rowH = (lane >> 4) << 3;
#pragma unroll
  for (int i = 0; i < 2; ++i) {
    const int mbase = mb0 + 16 * i + rowH;
#pragma unroll
    for (int j = 0; j < 4; ++j) {
      const int col = nb0 + 16 * j + colL;
      if (col >= Nlog) continue;
      const float bv = bias[col];
#pragma unroll
      for (int e = 0; e < 8; ++e) {
        const int row = mbase + e;
        if (row >= Mp) continue;
        float v = acc[i][j][e] + bv;
        if constexpr (MODE == 1) v += resid[(size_t)row * Nlog + col];
        if constexpr (MODE == 2) {
          v = 0.5f * v * (1.f + erff(v * 0.70710678118654752f));
          outb[(size_t)row * Nlog + col] = (bf16)v;
        } else {
          outf[(size_t)row * Nlog + col] = v;
        }
      }
    }
  }
}

// ---------------------------------------------------------------------------
// LayerNorm over D=768, one row per block (256 threads, 3 cols/thread), bf16 out
// ---------------------------------------------------------------------------
__global__ __launch_bounds__(256) void layernorm_kernel(
    const float* __restrict__ X, const float* __restrict__ g,
    const float* __restrict__ b, bf16* __restrict__ out, int rows)
{
  __shared__ float sdata[256];
  const int row = blockIdx.x;
  if (row >= rows) return;
  const float* x = X + (size_t)row * kD;
  const int tid = threadIdx.x;

  float xr[3];
  float s = 0.f;
#pragma unroll
  for (int i = 0; i < 3; ++i) { xr[i] = x[tid + 256 * i]; s += xr[i]; }
  sdata[tid] = s;
  __syncthreads();
  for (int st = 128; st > 0; st >>= 1) {
    if (tid < st) sdata[tid] += sdata[tid + st];
    __syncthreads();
  }
  const float mu = sdata[0] * (1.f / 768.f);
  __syncthreads();

  float v = 0.f;
#pragma unroll
  for (int i = 0; i < 3; ++i) { const float d = xr[i] - mu; v += d * d; }
  sdata[tid] = v;
  __syncthreads();
  for (int st = 128; st > 0; st >>= 1) {
    if (tid < st) sdata[tid] += sdata[tid + st];
    __syncthreads();
  }
  const float rs = rsqrtf(sdata[0] * (1.f / 768.f) + 1e-5f);

  bf16* o = out + (size_t)row * kD;
#pragma unroll
  for (int i = 0; i < 3; ++i) {
    const int c = tid + 256 * i;
    o[c] = (bf16)((xr[i] - mu) * rs * g[c] + b[c]);
  }
}

// ---------------------------------------------------------------------------
// Attention kernels. qkv: (B*N1, 2304) fp32, logical (b, t, {q|k|v}, head, 64).
// out: (B*N1, 768) bf16 at [.., head*64 + d]. Scale hd^-0.5 = 0.125 folded in.
// Wave-per-query; scores in LDS; lanes own 2 output dims in the AV pass.
// ---------------------------------------------------------------------------
DEV_ float wave_max(float m) {
#pragma unroll
  for (int o = 16; o; o >>= 1) m = fmaxf(m, __shfl_xor(m, o, 32));
  return m;
}
DEV_ float wave_sum(float s) {
#pragma unroll
  for (int o = 16; o; o >>= 1) s += __shfl_xor(s, o, 32);
  return s;
}

// Time attention: group (b, head, pi), queries over frames (L = 1 + F = 9).
__global__ __launch_bounds__(256) void attn_time_kernel(
    const float* __restrict__ qkv, bf16* __restrict__ outb)
{
  constexpr int L = 1 + kF;  // 9
  __shared__ float sh[8 * (64 + L)];
  const int wave = threadIdx.x >> 5, lane = threadIdx.x & 31;
  float* qs = sh + wave * (64 + L);
  float* sc = qs + 64;

  int qid = blockIdx.x * 8 + wave;       // B*H*n*F = 75264 queries
  const int fi = qid % kF; qid /= kF;
  const int pi = qid % kNP; qid /= kNP;
  const int head = qid % kH;
  const int b = qid / kH;
  const int tq = 1 + fi * kNP + pi;

  const float* qb = qkv + (size_t)(b * kN1 + tq) * 2304 + head * 64;
  for (int d = lane; d < 64; d += 32) qs[d] = qb[d] * 0.125f;
  __syncthreads();

  if (lane < L) {
    const int tk = (lane == 0) ? 0 : (1 + (lane - 1) * kNP + pi);
    const float* kb = qkv + (size_t)(b * kN1 + tk) * 2304 + 768 + head * 64;
    float s = 0.f;
    for (int d = 0; d < 64; ++d) s += qs[d] * kb[d];
    sc[lane] = s;
  }
  __syncthreads();

  float m = (lane < L) ? sc[lane] : -1e30f;
  m = wave_max(m);
  float sum = 0.f;
  if (lane < L) { const float p = __expf(sc[lane] - m); sc[lane] = p; sum = p; }
  sum = wave_sum(sum);
  __syncthreads();

  const float inv = 1.f / sum;
  const int d0 = lane * 2;
  float a0 = 0.f, a1 = 0.f;
  for (int j = 0; j < L; ++j) {
    const int tk = (j == 0) ? 0 : (1 + (j - 1) * kNP + pi);
    const float* vb = qkv + (size_t)(b * kN1 + tk) * 2304 + 1536 + head * 64;
    const float p = sc[j];
    a0 += p * vb[d0];
    a1 += p * vb[d0 + 1];
  }
  bf16* ob = outb + (size_t)(b * kN1 + tq) * kD + head * 64;
  ob[d0]     = (bf16)(a0 * inv);
  ob[d0 + 1] = (bf16)(a1 * inv);
}

// Spatial attention: group (b, head, fi), queries over patches (L = 1 + n = 197).
__global__ __launch_bounds__(256) void attn_spatial_kernel(
    const float* __restrict__ qkv, bf16* __restrict__ outb)
{
  constexpr int L = 1 + kNP;  // 197
  __shared__ float sh[8 * (64 + L)];
  const int wave = threadIdx.x >> 5, lane = threadIdx.x & 31;
  float* qs = sh + wave * (64 + L);
  float* sc = qs + 64;

  int qid = blockIdx.x * 8 + wave;       // B*H*F*n = 75264 queries
  const int qi = qid % kNP; qid /= kNP;
  const int fi = qid % kF; qid /= kF;
  const int head = qid % kH;
  const int b = qid / kH;
  const int tq = 1 + fi * kNP + qi;

  const float* qb = qkv + (size_t)(b * kN1 + tq) * 2304 + head * 64;
  for (int d = lane; d < 64; d += 32) qs[d] = qb[d] * 0.125f;
  __syncthreads();

  for (int j = lane; j < L; j += 32) {
    const int tk = (j == 0) ? 0 : (1 + fi * kNP + (j - 1));
    const float* kb = qkv + (size_t)(b * kN1 + tk) * 2304 + 768 + head * 64;
    float s = 0.f;
    for (int d = 0; d < 64; ++d) s += qs[d] * kb[d];
    sc[j] = s;
  }
  __syncthreads();

  float m = -1e30f;
  for (int j = lane; j < L; j += 32) m = fmaxf(m, sc[j]);
  m = wave_max(m);
  float sum = 0.f;
  for (int j = lane; j < L; j += 32) { const float p = __expf(sc[j] - m); sc[j] = p; sum += p; }
  sum = wave_sum(sum);
  __syncthreads();

  const float inv = 1.f / sum;
  const int d0 = lane * 2;
  float a0 = 0.f, a1 = 0.f;
  for (int j = 0; j < L; ++j) {
    const int tk = (j == 0) ? 0 : (1 + fi * kNP + (j - 1));
    const float* vb = qkv + (size_t)(b * kN1 + tk) * 2304 + 1536 + head * 64;
    const float p = sc[j];
    a0 += p * vb[d0];
    a1 += p * vb[d0 + 1];
  }
  bf16* ob = outb + (size_t)(b * kN1 + tq) * kD + head * 64;
  ob[d0]     = (bf16)(a0 * inv);
  ob[d0 + 1] = (bf16)(a1 * inv);
}

// CLS attention: one wave per (b, head); query = token 0 over all N1 tokens.
__global__ __launch_bounds__(32) void attn_cls_kernel(
    const float* __restrict__ qkv, bf16* __restrict__ outb)
{
  __shared__ float qs[64];
  __shared__ float sc[kN1];
  const int lane = threadIdx.x;
  const int head = blockIdx.x % kH;
  const int b = blockIdx.x / kH;

  const float* qb = qkv + (size_t)(b * kN1) * 2304 + head * 64;
  for (int d = lane; d < 64; d += 32) qs[d] = qb[d] * 0.125f;
  __syncthreads();

  for (int j = lane; j < kN1; j += 32) {
    const float* kb = qkv + (size_t)(b * kN1 + j) * 2304 + 768 + head * 64;
    float s = 0.f;
    for (int d = 0; d < 64; ++d) s += qs[d] * kb[d];
    sc[j] = s;
  }
  __syncthreads();

  float m = -1e30f;
  for (int j = lane; j < kN1; j += 32) m = fmaxf(m, sc[j]);
  m = wave_max(m);
  float sum = 0.f;
  for (int j = lane; j < kN1; j += 32) { const float p = __expf(sc[j] - m); sc[j] = p; sum += p; }
  sum = wave_sum(sum);
  __syncthreads();

  const float inv = 1.f / sum;
  const int d0 = lane * 2;
  float a0 = 0.f, a1 = 0.f;
  for (int j = 0; j < kN1; ++j) {
    const float* vb = qkv + (size_t)(b * kN1 + j) * 2304 + 1536 + head * 64;
    const float p = sc[j];
    a0 += p * vb[d0];
    a1 += p * vb[d0 + 1];
  }
  bf16* ob = outb + (size_t)(b * kN1) * kD + head * 64;
  ob[d0]     = (bf16)(a0 * inv);
  ob[d0 + 1] = (bf16)(a1 * inv);
}

// ---------------------------------------------------------------------------
// Small helper kernels
// ---------------------------------------------------------------------------
__global__ void cvt_bf16_kernel(const float* __restrict__ src, bf16* __restrict__ dst,
                                long nsrc, long ndst)
{
  const long i = (long)blockIdx.x * blockDim.x + threadIdx.x;
  if (i < ndst) dst[i] = (i < nsrc) ? (bf16)src[i] : (bf16)0.f;
}

// im2col: video (B,F,3,224,224) -> Xp (B*F*196, 768) with k = c*256 + ph*16 + pw
__global__ void im2col_kernel(const float* __restrict__ video, bf16* __restrict__ xp)
{
  const long total = (long)kPT * kD;
  const long idx = (long)blockIdx.x * blockDim.x + threadIdx.x;
  if (idx >= total) return;
  const int kcol = (int)(idx % kD);
  long rowid = idx / kD;
  const int pi = (int)(rowid % kNP);
  const int bf = (int)(rowid / kNP);     // 0..31 (b*F + fi)
  const int c  = kcol >> 8;
  const int ph = (kcol >> 4) & 15;
  const int pw = kcol & 15;
  const int hp = pi / 14, wp = pi % 14;
  const long voff = ((long)bf * 3 + c) * 224 * 224 + (long)(hp * 16 + ph) * 224 + (wp * 16 + pw);
  xp[idx] = (bf16)video[voff];
}

// Assemble x (T,768): cls token + positional/temporal embeddings + patch tokens
__global__ __launch_bounds__(256) void assemble_x_kernel(
    const float* __restrict__ ptok, const float* __restrict__ cls_tok,
    const float* __restrict__ pos, const float* __restrict__ temp,
    float* __restrict__ X)
{
  const int row = blockIdx.x;            // 0..T-1
  const int b = row / kN1, t = row % kN1;
  float* xr = X + (size_t)row * kD;
  if (t == 0) {
    for (int c = threadIdx.x; c < kD; c += 256)
      xr[c] = cls_tok[c] + pos[c];
  } else {
    const int tt = t - 1, fi = tt / kNP, pi = tt % kNP;
    const float* pr = ptok + (size_t)((b * kF + fi) * kNP + pi) * kD;
    const float* pe = pos + (size_t)(1 + pi) * kD;
    const float* te = temp + (size_t)fi * kD;
    for (int c = threadIdx.x; c < kD; c += 256)
      xr[c] = pr[c] + pe[c] + te[c];
  }
}

// Gather 4 cls rows (bf16) into a 16x768 tile (rows 4..15 zero)
__global__ void gather_cls_kernel(const bf16* __restrict__ xbf, bf16* __restrict__ cls)
{
  const int i = blockIdx.x * blockDim.x + threadIdx.x;
  if (i >= 16 * kD) return;
  const int r = i / kD, c = i % kD;
  cls[i] = (r < kB) ? xbf[(size_t)(r * kN1) * kD + c] : (bf16)0.f;
}

__global__ void copy_out_kernel(const float* __restrict__ hout, float* __restrict__ out)
{
  const int i = blockIdx.x * blockDim.x + threadIdx.x;
  if (i < kB * 1000) out[i] = hout[i];   // hout rows 0..3 contiguous, ld=1000
}

// ---------------------------------------------------------------------------
// Host-side launchers
// ---------------------------------------------------------------------------
static inline void launch_gemm(hipStream_t s, int mode,
                               const bf16* A, const bf16* W,
                               const float* bias, const float* resid,
                               float* outf, bf16* outb,
                               int Mp, int Npad, int Nlog, int K)
{
  dim3 grid((Npad + 255) / 256, (Mp + 63) / 64);
  if (mode == 0)
    wmma_gemm_t<0><<<grid, dim3(256), 0, s>>>(A, W, bias, resid, outf, outb, Mp, Npad, Nlog, K);
  else if (mode == 1)
    wmma_gemm_t<1><<<grid, dim3(256), 0, s>>>(A, W, bias, resid, outf, outb, Mp, Npad, Nlog, K);
  else
    wmma_gemm_t<2><<<grid, dim3(256), 0, s>>>(A, W, bias, resid, outf, outb, Mp, Npad, Nlog, K);
}

static inline void launch_cvt(hipStream_t s, const float* src, bf16* dst,
                              long nsrc, long ndst)
{
  const long blocks = (ndst + 255) / 256;
  cvt_bf16_kernel<<<dim3((unsigned)blocks), dim3(256), 0, s>>>(src, dst, nsrc, ndst);
}

static inline size_t alignup(size_t x) { return (x + 255) & ~(size_t)255; }

extern "C" void kernel_launch(void* const* d_in, const int* in_sizes, int n_in,
                              void* d_out, int out_size, void* d_ws, size_t ws_size,
                              hipStream_t stream)
{
  (void)in_sizes; (void)n_in; (void)out_size; (void)ws_size;

  const float* video   = (const float*)d_in[0];
  const float* cls_tok = (const float*)d_in[1];
  const float* pos     = (const float*)d_in[2];
  const float* temb    = (const float*)d_in[3];
  const float* patch_w = (const float*)d_in[4];
  const float* patch_b = (const float*)d_in[5];
  const float* ln1_g   = (const float*)d_in[6];
  const float* ln1_b   = (const float*)d_in[7];
  const float* aqkv_w  = (const float*)d_in[8];
  const float* aqkv_b  = (const float*)d_in[9];
  const float* aproj_w = (const float*)d_in[10];
  const float* aproj_b = (const float*)d_in[11];
  const float* ln3_g   = (const float*)d_in[12];
  const float* ln3_b   = (const float*)d_in[13];
  const float* tqkv_w  = (const float*)d_in[14];
  const float* tqkv_b  = (const float*)d_in[15];
  const float* tproj_w = (const float*)d_in[16];
  const float* tproj_b = (const float*)d_in[17];
  const float* ln2_g   = (const float*)d_in[18];
  const float* ln2_b   = (const float*)d_in[19];
  const float* mlp_w1  = (const float*)d_in[20];
  const float* mlp_b1  = (const float*)d_in[21];
  const float* mlp_w2  = (const float*)d_in[22];
  const float* mlp_b2  = (const float*)d_in[23];
  const float* norm_g  = (const float*)d_in[24];
  const float* norm_b  = (const float*)d_in[25];
  const float* head_w  = (const float*)d_in[26];
  const float* head_b  = (const float*)d_in[27];

  // ---------------- workspace layout ----------------
  char* ws = (char*)d_ws;
  size_t off = 0;
  auto walloc = [&](size_t bytes) -> void* {
    void* p = ws + off;
    off = alignup(off + bytes);
    return p;
  };
  const size_t WBF_ELEMS = SZ_PATCHW + 4 * SZ_LAYER + SZ_HEADP;  // 39,112,704
  bf16*  WBF   = (bf16*)walloc(WBF_ELEMS * sizeof(bf16));
  bf16*  XP    = (bf16*)walloc((size_t)kPT * kD * sizeof(bf16));
  float* S1    = (float*)walloc((size_t)kTp * 2304 * sizeof(float)); // qkv / patch-gemm out
  float* X     = (float*)walloc((size_t)kTp * kD * sizeof(float));
  float* Y1    = (float*)walloc((size_t)kTp * kD * sizeof(float));   // tr
  float* Y2    = (float*)walloc((size_t)kTp * kD * sizeof(float));   // sr
  bf16*  XBF   = (bf16*)walloc((size_t)kTp * kD * sizeof(bf16));     // ln outputs
  bf16*  HBF   = (bf16*)walloc((size_t)kTp * kDF * sizeof(bf16));    // mlp hidden
  bf16*  ABF   = (bf16*)walloc((size_t)kTp * kD * sizeof(bf16));     // attn output
  bf16*  CLSBF = (bf16*)walloc((size_t)16 * kD * sizeof(bf16));
  float* HOUT  = (float*)walloc((size_t)16 * 1000 * sizeof(float));

  // bf16 weight sub-pointers
  bf16* W_patch = WBF;
  bf16* W_head  = WBF + SZ_PATCHW + 4 * SZ_LAYER;

  // ---------------- convert weights to bf16 ----------------
  launch_cvt(stream, patch_w, W_patch, (long)SZ_PATCHW, (long)SZ_PATCHW);
  for (int i = 0; i < 4; ++i) {
    bf16* base = WBF + SZ_PATCHW + (size_t)i * SZ_LAYER;
    launch_cvt(stream, tqkv_w  + (size_t)i * SZ_QKVW,  base,                              (long)SZ_QKVW,  (long)SZ_QKVW);
    launch_cvt(stream, tproj_w + (size_t)i * SZ_PROJW, base + SZ_QKVW,                    (long)SZ_PROJW, (long)SZ_PROJW);
    launch_cvt(stream, aqkv_w  + (size_t)i * SZ_QKVW,  base + SZ_QKVW + SZ_PROJW,         (long)SZ_QKVW,  (long)SZ_QKVW);
    launch_cvt(stream, aproj_w + (size_t)i * SZ_PROJW, base + 2 * SZ_QKVW + SZ_PROJW,     (long)SZ_PROJW, (long)SZ_PROJW);
    launch_cvt(stream, mlp_w1  + (size_t)i * SZ_W1,    base + 2 * SZ_QKVW + 2 * SZ_PROJW, (long)SZ_W1,    (long)SZ_W1);
    launch_cvt(stream, mlp_w2  + (size_t)i * SZ_W2,    base + 2 * SZ_QKVW + 2 * SZ_PROJW + SZ_W1,
               (long)SZ_W2, (long)SZ_W2);
  }
  launch_cvt(stream, head_w, W_head, (long)1000 * 768, (long)SZ_HEADP); // zero-pads rows 1000..1007

  // ---------------- patch embed + token assembly ----------------
  {
    const long total = (long)kPT * kD;
    im2col_kernel<<<dim3((unsigned)((total + 255) / 256)), dim3(256), 0, stream>>>(video, XP);
    launch_gemm(stream, 0, XP, W_patch, patch_b, nullptr, S1, nullptr,
                kPT, kD, kD, kD);
    assemble_x_kernel<<<dim3(kT), dim3(256), 0, stream>>>(S1, cls_tok, pos, temb, X);
  }

  // ---------------- transformer layers ----------------
  const int attn_grid = (kB * kH * kF * kNP) / 8;  // 9408 blocks, 8 waves each
  for (int i = 0; i < 4; ++i) {
    bf16* base    = WBF + SZ_PATCHW + (size_t)i * SZ_LAYER;
    bf16* Wb_tqkv = base;
    bf16* Wb_tprj = base + SZ_QKVW;
    bf16* Wb_aqkv = base + SZ_QKVW + SZ_PROJW;
    bf16* Wb_aprj = base + 2 * SZ_QKVW + SZ_PROJW;
    bf16* Wb_w1   = base + 2 * SZ_QKVW + 2 * SZ_PROJW;
    bf16* Wb_w2   = Wb_w1 + SZ_W1;

    // ---- time attention branch: a = LN3(x); t = attn_time(a); tr = x + t
    layernorm_kernel<<<dim3(kTp), dim3(256), 0, stream>>>(X, ln3_g + i * kD, ln3_b + i * kD, XBF, kTp);
    launch_gemm(stream, 0, XBF, Wb_tqkv, tqkv_b + (size_t)i * 2304, nullptr, S1, nullptr,
                kTp, 2304, 2304, kD);
    attn_time_kernel<<<dim3(attn_grid), dim3(256), 0, stream>>>(S1, ABF);
    attn_cls_kernel<<<dim3(kB * kH), dim3(32), 0, stream>>>(S1, ABF);
    launch_gemm(stream, 1, ABF, Wb_tprj, tproj_b + (size_t)i * kD, X, Y1, nullptr,
                kTp, kD, kD, kD);

    // ---- spatial attention branch: s = attn_space(LN1(tr)); sr = x + s
    layernorm_kernel<<<dim3(kTp), dim3(256), 0, stream>>>(Y1, ln1_g + i * kD, ln1_b + i * kD, XBF, kTp);
    launch_gemm(stream, 0, XBF, Wb_aqkv, aqkv_b + (size_t)i * 2304, nullptr, S1, nullptr,
                kTp, 2304, 2304, kD);
    attn_spatial_kernel<<<dim3(attn_grid), dim3(256), 0, stream>>>(S1, ABF);
    attn_cls_kernel<<<dim3(kB * kH), dim3(32), 0, stream>>>(S1, ABF);
    launch_gemm(stream, 1, ABF, Wb_aprj, aproj_b + (size_t)i * kD, X, Y2, nullptr,
                kTp, kD, kD, kD);

    // ---- MLP: x = sr + W2(gelu(W1(LN2(sr))))
    layernorm_kernel<<<dim3(kTp), dim3(256), 0, stream>>>(Y2, ln2_g + i * kD, ln2_b + i * kD, XBF, kTp);
    launch_gemm(stream, 2, XBF, Wb_w1, mlp_b1 + (size_t)i * kDF, nullptr, nullptr, HBF,
                kTp, kDF, kDF, kD);
    launch_gemm(stream, 1, HBF, Wb_w2, mlp_b2 + (size_t)i * kD, Y2, X, nullptr,
                kTp, kD, kD, kDF);
  }

  // ---------------- final norm + head ----------------
  layernorm_kernel<<<dim3(kTp), dim3(256), 0, stream>>>(X, norm_g, norm_b, XBF, kTp);
  gather_cls_kernel<<<dim3((16 * kD + 255) / 256), dim3(256), 0, stream>>>(XBF, CLSBF);
  launch_gemm(stream, 0, CLSBF, W_head, head_b, nullptr, HOUT, nullptr,
              16, 1008, 1000, kD);
  copy_out_kernel<<<dim3((kB * 1000 + 255) / 256), dim3(256), 0, stream>>>(HOUT, (float*)d_out);
}
